// GATv2ConvLayer_67379446940468
// MI455X (gfx1250) — matrix operand order repaired and checked by reading, not compiled, not run on hardware
//
#include <hip/hip_runtime.h>
#include <hip/hip_bf16.h>

#define DIM 64
#define NEG_SLOPE 0.2f

typedef __attribute__((ext_vector_type(16))) __bf16 v16bf;
typedef __attribute__((ext_vector_type(8)))  float  v8f;

// ---------- helpers ----------
__device__ __forceinline__ unsigned encF(float f) {
    unsigned u = __float_as_uint(f);
    return (u & 0x80000000u) ? ~u : (u | 0x80000000u);
}
__device__ __forceinline__ float decF(unsigned u) {
    return __uint_as_float((u & 0x80000000u) ? (u & 0x7FFFFFFFu) : ~u);
}

// ---------- kernel 1: dual GEMM  xl = x@Wl + bl,  xr = x@Wr + br ----------
// One wave per 16-row stripe; K=64 -> two 16x16x32 bf16 WMMAs per 16x16 tile;
// 4 column tiles per weight, 2 weights => 16 v_wmma per wave.
__global__ __launch_bounds__(256)
void k_gemm_dual(const float* __restrict__ x,
                 const float* __restrict__ Wl, const float* __restrict__ bl,
                 const float* __restrict__ Wr, const float* __restrict__ br,
                 float* __restrict__ xl, float* __restrict__ xr, int N)
{
    const int wave = threadIdx.x >> 5;
    const int lane = threadIdx.x & 31;
    const int rowTile = blockIdx.x * 8 + wave;
    const int row0 = rowTile * 16;
    if (row0 >= N) return;

    const int m    = lane & 15;   // row within tile (A), col within tile (B/C)
    const int half = lane >> 4;   // K-group selector
    const bool tileFull = (row0 + 16 <= N);   // uniform across the wave

    int r = row0 + m;
    if (r >= N) r = N - 1;        // clamp for ragged tail (stores guarded)
    const float* __restrict__ xrow = x + (size_t)r * DIM;

    // A fragments: 16x32 bf16.  lane m, half h: elems 0..15 = K h*16..h*16+15
    // (same k order as B).
    v16bf a0, a1;
    const int kb = half * 16;
#pragma unroll
    for (int e = 0; e < 8; ++e) {
        a0[e]     = (__bf16)xrow[kb + e];
        a0[e + 8] = (__bf16)xrow[kb + 8 + e];
        a1[e]     = (__bf16)xrow[32 + kb + e];
        a1[e + 8] = (__bf16)xrow[32 + kb + 8 + e];
    }

#pragma unroll
    for (int w = 0; w < 2; ++w) {
        const float* __restrict__ W  = (w == 0) ? Wl : Wr;
        const float* __restrict__ bs = (w == 0) ? bl : br;
        float* __restrict__ out      = (w == 0) ? xl : xr;

#pragma unroll
        for (int ct = 0; ct < 4; ++ct) {
            const int n0 = ct * 16;
            const int n  = n0 + m;
            // B fragments: 32x16 bf16, B[k][n] = W[k*DIM + n].
            // lane m = col, half h: elems 0..15 = K h*16 .. h*16+15.
            v16bf b0, b1;
            const int kbB = half * 16;
#pragma unroll
            for (int e = 0; e < 16; ++e) {
                b0[e] = (__bf16)W[(size_t)(kbB + e) * DIM + n];
                b1[e] = (__bf16)W[(size_t)(32 + kbB + e) * DIM + n];
            }
            // C init with bias broadcast down the column
            v8f c;
            const float bv = bs[n];
#pragma unroll
            for (int i = 0; i < 8; ++i) c[i] = bv;

            c = __builtin_amdgcn_wmma_f32_16x16x32_bf16(
                    false, a0, false, b0, (short)0, c, false, false);
            c = __builtin_amdgcn_wmma_f32_16x16x32_bf16(
                    false, a1, false, b1, (short)0, c, false, false);

            // C/D layout: element i -> row (i + 8*half), col (lane&15)
            float* __restrict__ base = out + (size_t)(row0 + 8 * half) * DIM + n0 + m;
            if (tileFull) {
#pragma unroll
                for (int i = 0; i < 8; ++i)
                    base[(size_t)i * DIM] = c[i];
            } else {
#pragma unroll
                for (int i = 0; i < 8; ++i) {
                    if (row0 + 8 * half + i < N)
                        base[(size_t)i * DIM] = c[i];
                }
            }
        }
    }
}

// ---------- kernel 2: zero a float region (float4) ----------
__global__ __launch_bounds__(256)
void k_zero4(float4* __restrict__ p, int n4)
{
    int i = blockIdx.x * 256 + threadIdx.x;
    if (i < n4) p[i] = make_float4(0.f, 0.f, 0.f, 0.f);
}

// ---------- kernel 3: per-edge score + segment max (wave per edge) ----------
__global__ __launch_bounds__(256)
void k_edge_score(const float* __restrict__ xl, const float* __restrict__ xr,
                  const int* __restrict__ src, const int* __restrict__ dst,
                  const float* __restrict__ att,
                  float* __restrict__ e, unsigned* __restrict__ mEnc, int E)
{
    const int gw   = (blockIdx.x * 256 + threadIdx.x) >> 5;
    const int lane = threadIdx.x & 31;
    if (gw >= E) return;
    const int s = src[gw], d = dst[gw];

    const float2 vl = *(const float2*)(xl + (size_t)s * DIM + lane * 2);
    const float2 vr = *(const float2*)(xr + (size_t)d * DIM + lane * 2);
    const float2 av = *(const float2*)(att + lane * 2);

    float h0 = vl.x + vr.x; h0 = (h0 > 0.0f) ? h0 : NEG_SLOPE * h0;
    float h1 = vl.y + vr.y; h1 = (h1 > 0.0f) ? h1 : NEG_SLOPE * h1;
    float p = h0 * av.x + h1 * av.y;

#pragma unroll
    for (int off = 16; off > 0; off >>= 1)
        p += __shfl_xor(p, off, 32);

    if (lane == 0) {
        e[gw] = p;
        atomicMax(&mEnc[d], encF(p));
    }
}

// ---------- kernel 4: exp + denom (thread per edge) ----------
__global__ __launch_bounds__(256)
void k_edge_exp(const int* __restrict__ dst, const unsigned* __restrict__ mEnc,
                float* __restrict__ e, float* __restrict__ denom, int E)
{
    int k = blockIdx.x * 256 + threadIdx.x;
    if (k >= E) return;
    const int d = dst[k];
    const float ex = __expf(e[k] - decF(mEnc[d]));
    e[k] = ex;
    atomicAdd(&denom[d], ex);
}

// ---------- kernel 5: weighted scatter (wave per edge) ----------
__global__ __launch_bounds__(256)
void k_edge_scatter(const float* __restrict__ xl,
                    const int* __restrict__ src, const int* __restrict__ dst,
                    const float* __restrict__ e, const float* __restrict__ denom,
                    float* __restrict__ acc, int E)
{
    const int gw   = (blockIdx.x * 256 + threadIdx.x) >> 5;
    const int lane = threadIdx.x & 31;
    if (gw >= E) return;
    const int s = src[gw], d = dst[gw];
    const float alpha = e[gw] / denom[d];

    const float2 vl = *(const float2*)(xl + (size_t)s * DIM + lane * 2);
    float* a = acc + (size_t)d * DIM + lane * 2;
    atomicAdd(a + 0, alpha * vl.x);
    atomicAdd(a + 1, alpha * vl.y);
}

// ---------- kernel 6: epilogue  out = acc + bias (+ resid), float4 ----------
__global__ __launch_bounds__(256)
void k_epilogue4(const float4* __restrict__ acc, const float4* __restrict__ bias4,
                 const float4* __restrict__ resid, float4* __restrict__ out, int n4)
{
    int i = blockIdx.x * 256 + threadIdx.x;
    if (i >= n4) return;
    const float4 a = acc[i];
    const float4 b = bias4[i & (DIM / 4 - 1)];
    float4 v = make_float4(a.x + b.x, a.y + b.y, a.z + b.z, a.w + b.w);
    if (resid) {
        const float4 rr = resid[i];
        v.x += rr.x; v.y += rr.y; v.z += rr.z; v.w += rr.w;
    }
    out[i] = v;
}

// ---------- host-side orchestration ----------
extern "C" void kernel_launch(void* const* d_in, const int* in_sizes, int n_in,
                              void* d_out, int out_size, void* d_ws, size_t ws_size,
                              hipStream_t stream)
{
    const float* x    = (const float*)d_in[0];
    const int*   ei   = (const int*)  d_in[1];
    const float* Wl   = (const float*)d_in[2];
    const float* bl   = (const float*)d_in[3];
    const float* Wr   = (const float*)d_in[4];
    const float* br   = (const float*)d_in[5];
    const float* att  = (const float*)d_in[6];
    const float* bias = (const float*)d_in[7];

    const int N = in_sizes[0] / DIM;
    const int E = in_sizes[1] / 2;
    const int* src = ei;
    const int* dst = ei + E;

    // workspace layout (floats)
    float* ws = (float*)d_ws;
    const size_t nd = (size_t)N * DIM;
    float*    xl    = ws;
    float*    xr    = ws + nd;
    float*    h1    = ws + 2 * nd;
    float*    acc   = ws + 3 * nd;          // acc | mEnc | denom contiguous
    unsigned* mEnc  = (unsigned*)(ws + 4 * nd);
    float*    denom = ws + 4 * nd + N;
    float*    ebuf  = ws + 4 * nd + 2 * N;

    const int zero4      = (int)((nd + 2 * (size_t)N) / 4);  // acc+mEnc+denom (nd,N mult of 4... N*2/4 ok: nd mult of 64, 2N mult of 2)
    const int zero4c     = (int)((nd + 2 * (size_t)N + 3) / 4);
    const int gemmBlocks = ((N + 15) / 16 + 7) / 8;
    const int waveBlocks = (E + 7) / 8;                 // 8 waves (edges) / block
    const int thrBlocksE = (E + 255) / 256;
    const int n4         = (int)(nd / 4);
    const int elemBlocks = (n4 + 255) / 256;
    const int zeroBlocks = (zero4c + 255) / 256;
    (void)zero4;

    for (int layer = 0; layer < 2; ++layer) {
        const float* hin = (layer == 0) ? x : h1;

        k_gemm_dual<<<gemmBlocks, 256, 0, stream>>>(hin, Wl, bl, Wr, br, xl, xr, N);
        k_zero4<<<zeroBlocks, 256, 0, stream>>>((float4*)acc, zero4c);
        k_edge_score<<<waveBlocks, 256, 0, stream>>>(xl, xr, src, dst, att, ebuf, mEnc, E);
        k_edge_exp<<<thrBlocksE, 256, 0, stream>>>(dst, mEnc, ebuf, denom, E);
        k_edge_scatter<<<waveBlocks, 256, 0, stream>>>(xl, src, dst, ebuf, denom, acc, E);

        if (layer == 0) {
            k_epilogue4<<<elemBlocks, 256, 0, stream>>>((const float4*)acc, (const float4*)bias,
                                                        nullptr, (float4*)h1, n4);
        } else {
            k_epilogue4<<<elemBlocks, 256, 0, stream>>>((const float4*)acc, (const float4*)bias,
                                                        (const float4*)x, (float4*)d_out, n4);
        }
    }
}
